// GLCN_21758304321775
// MI455X (gfx1250) — matrix operand order repaired and checked by reading, not compiled, not run on hardware
//
#include <hip/hip_runtime.h>
#include <cstdint>
#include <cstddef>

#define GN 2048
#define GF 512
#define GD 512

typedef __attribute__((ext_vector_type(16))) __bf16 v16bf;
typedef __attribute__((ext_vector_type(8)))  __bf16 v8bf;
typedef __attribute__((ext_vector_type(8)))  float  v8f;

__device__ __forceinline__ unsigned short f32_to_bf16(float f) {
  unsigned int u = __float_as_uint(f);
  u += 0x7FFFu + ((u >> 16) & 1u);            // round-to-nearest-even
  return (unsigned short)(u >> 16);
}

// ---------------- elementwise helpers ----------------
__global__ __launch_bounds__(256) void cvt_bf16_kernel(const float* __restrict__ in,
                                                       unsigned short* __restrict__ out, int n) {
  int i = blockIdx.x * 256 + threadIdx.x;
  if (i < n) out[i] = f32_to_bf16(in[i]);
}

__global__ __launch_bounds__(256) void copy_f32_kernel(const float* __restrict__ in,
                                                       float* __restrict__ out, int n) {
  int i = blockIdx.x * 256 + threadIdx.x;
  if (i < n) out[i] = in[i];
}

// ---------------- link prediction: Aadj[i,j] ----------------
__global__ __launch_bounds__(256) void linkpred_kernel(const float* __restrict__ X,
                                                       const float* __restrict__ gumbel,
                                                       const float* __restrict__ a_link,
                                                       float* __restrict__ Aout) {
  __shared__ float hI[64][65];
  __shared__ float hJ[64][65];
  __shared__ float aS[64];
  const int tid = threadIdx.x;
  const int i0 = blockIdx.y * 64;
  const int j0 = blockIdx.x * 64;
  for (int idx = tid; idx < 64 * 64; idx += 256) {
    int r = idx >> 6, c = idx & 63;
    hI[r][c] = X[(size_t)(i0 + r) * GF + c];
    hJ[r][c] = X[(size_t)(j0 + r) * GF + c];
  }
  if (tid < 64) aS[tid] = a_link[tid];
  __syncthreads();

  const int ti = tid >> 4, tj = tid & 15;
  float s[4][4];
  #pragma unroll
  for (int ii = 0; ii < 4; ++ii)
    #pragma unroll
    for (int jj = 0; jj < 4; ++jj) s[ii][jj] = 0.0f;

  for (int k = 0; k < 64; ++k) {
    float ak = aS[k];
    float hi[4], hj[4];
    #pragma unroll
    for (int ii = 0; ii < 4; ++ii) hi[ii] = hI[ti * 4 + ii][k];
    #pragma unroll
    for (int jj = 0; jj < 4; ++jj) hj[jj] = hJ[tj * 4 + jj][k];
    #pragma unroll
    for (int ii = 0; ii < 4; ++ii)
      #pragma unroll
      for (int jj = 0; jj < 4; ++jj)
        s[ii][jj] += fabsf(hi[ii] - hj[jj]) * ak;
  }

  const float invtau = 1.0f / 1.4f;
  #pragma unroll
  for (int ii = 0; ii < 4; ++ii) {
    #pragma unroll
    for (int jj = 0; jj < 4; ++jj) {
      int i = i0 + ti * 4 + ii;
      int j = j0 + tj * 4 + jj;
      float g = gumbel[(size_t)i * GN + j];
      float z = (s[ii][jj] + g) * invtau;
      float ys = 1.0f / (1.0f + __expf(-z));
      float yh = (ys > 0.5f) ? 1.0f : 0.0f;
      float a = (yh - ys) + ys;                 // straight-through, forward value
      if (i == j) a = 1.0f;
      Aout[(size_t)i * GN + j] = a;
    }
  }
}

// ---------------- bf16 WMMA GEMM: C[M,N] = A[M,K] * B[K,N] ----------------
// Double-buffered LDS; A tile staged with GLOBAL_LOAD_ASYNC_TO_LDS_B128 (ASYNCcnt),
// B tile staged via registers with an in-LDS transpose to K-innermost layout.
// Optional fused bf16 copy of C in the epilogue (Cbf != nullptr).
#define MBLK 128
#define NBLK 128
#define KBLK 32
#define LSTR 40   // bf16 elements per LDS row (32 + 8 pad), 80B = 16B-aligned stride

__global__ __launch_bounds__(256) void gemm_bf16_wmma_kernel(const unsigned short* __restrict__ A,
                                                             const unsigned short* __restrict__ B,
                                                             float* __restrict__ C,
                                                             unsigned short* __restrict__ Cbf,
                                                             int M, int Nn, int K) {
  __shared__ __align__(16) unsigned short As[2][MBLK * LSTR]; // [m][k]
  __shared__ __align__(16) unsigned short Bs[2][NBLK * LSTR]; // [n][k] (K innermost)
  const int tid   = threadIdx.x;
  const int lane  = tid & 31;
  const int wave  = tid >> 5;      // 0..7
  const int waveM = wave >> 1;     // 0..3 -> 32-row slice
  const int waveN = wave & 1;      // 0..1 -> 64-col slice
  const int blockM = blockIdx.y * MBLK;
  const int blockN = blockIdx.x * NBLK;
  const int halfsel = lane >> 4;   // lanes 16-31
  const int l15     = lane & 15;
  const int aKoff = halfsel * 8;   // A frag: hi lanes shifted +8 in K
  const int bKoff = halfsel * 8;   // B frag: hi lanes shifted +8 in K

  v8f acc[2][4];
  #pragma unroll
  for (int mi = 0; mi < 2; ++mi)
    #pragma unroll
    for (int ni = 0; ni < 4; ++ni)
      acc[mi][ni] = (v8f){0.f, 0.f, 0.f, 0.f, 0.f, 0.f, 0.f, 0.f};

  const int ar = tid >> 1;           // A tile row 0..127
  const int ah = (tid & 1) * 16;     // A tile col half (bf16 elements)
  const int bk = tid >> 3;           // B tile K row 0..31
  const int bn = (tid & 7) * 16;     // B tile col base

  const unsigned short* Abase = A + (size_t)(blockM + ar) * K + ah;
  const unsigned short* Bbase = B + (size_t)bk * Nn + blockN + bn;

  auto stage = [&](int k0, int buf) {
    // --- A tile: async global -> LDS, 2 x 16B per thread, no VGPR round-trip ---
    unsigned lda = (unsigned)(size_t)(void*)&As[buf][ar * LSTR + ah];
    unsigned long long ga = (unsigned long long)(size_t)(Abase + k0);
    asm volatile("global_load_async_to_lds_b128 %0, %1, off"
                 :: "v"(lda), "v"(ga) : "memory");
    asm volatile("global_load_async_to_lds_b128 %0, %1, off offset:16"
                 :: "v"(lda), "v"(ga) : "memory");
    // --- B tile: coalesced b128 load, transposed scalar LDS stores ---
    const unsigned short* src = Bbase + (size_t)k0 * Nn;
    union { v8bf v; unsigned short u[8]; } x0, x1;
    x0.v = *reinterpret_cast<const v8bf*>(src);
    x1.v = *reinterpret_cast<const v8bf*>(src + 8);
    #pragma unroll
    for (int e = 0; e < 8; ++e) {
      Bs[buf][(bn + e)     * LSTR + bk] = x0.u[e];
      Bs[buf][(bn + 8 + e) * LSTR + bk] = x1.u[e];
    }
  };

  stage(0, 0);
  asm volatile("s_wait_asynccnt 0" ::: "memory");
  __syncthreads();

  int buf = 0;
  for (int k0 = 0; k0 < K; k0 += KBLK) {
    if (k0 + KBLK < K) stage(k0 + KBLK, buf ^ 1);

    union Frag { v16bf v; v8bf h[2]; };
    Frag af[2], bg[4];
    #pragma unroll
    for (int mi = 0; mi < 2; ++mi) {
      // 16-bit A 16x32 layout: lane holds row l15; K = (e<8?e:e+8) + 8*halfsel
      const unsigned short* ap = &As[buf][(waveM * 32 + mi * 16 + l15) * LSTR];
      af[mi].h[0] = *reinterpret_cast<const v8bf*>(ap + aKoff);
      af[mi].h[1] = *reinterpret_cast<const v8bf*>(ap + aKoff + 16);
    }
    #pragma unroll
    for (int ni = 0; ni < 4; ++ni) {
      // 16-bit B 32x16 layout: lane holds col l15; K = (e<8?e:e+8) + 8*halfsel
      const unsigned short* bp = &Bs[buf][(waveN * 64 + ni * 16 + l15) * LSTR];
      bg[ni].h[0] = *reinterpret_cast<const v8bf*>(bp + bKoff);
      bg[ni].h[1] = *reinterpret_cast<const v8bf*>(bp + bKoff + 16);
    }
    #pragma unroll
    for (int mi = 0; mi < 2; ++mi)
      #pragma unroll
      for (int ni = 0; ni < 4; ++ni)
        acc[mi][ni] = __builtin_amdgcn_wmma_f32_16x16x32_bf16(
            false, af[mi].v, false, bg[ni].v, (short)0, acc[mi][ni], false, false);

    // my async copies into buf^1 done; barrier: everyone done writing buf^1
    // and done consuming buf -> single barrier per K-step
    asm volatile("s_wait_asynccnt 0" ::: "memory");
    __syncthreads();
    buf ^= 1;
  }

  // C/D layout: VGPR r -> M = r + 8*halfsel, N = l15
  #pragma unroll
  for (int mi = 0; mi < 2; ++mi) {
    #pragma unroll
    for (int ni = 0; ni < 4; ++ni) {
      const int mBase = blockM + waveM * 32 + mi * 16 + halfsel * 8;
      const int col   = blockN + waveN * 64 + ni * 16 + l15;
      float* cp = C + (size_t)mBase * Nn + col;
      #pragma unroll
      for (int r = 0; r < 8; ++r) cp[(size_t)r * Nn] = acc[mi][ni][r];
    }
  }
  if (Cbf) {  // fused f32 -> bf16 copy of C (saves a full convert pass)
    #pragma unroll
    for (int mi = 0; mi < 2; ++mi) {
      #pragma unroll
      for (int ni = 0; ni < 4; ++ni) {
        const int mBase = blockM + waveM * 32 + mi * 16 + halfsel * 8;
        const int col   = blockN + waveN * 64 + ni * 16 + l15;
        unsigned short* bp = Cbf + (size_t)mBase * Nn + col;
        #pragma unroll
        for (int r = 0; r < 8; ++r) bp[(size_t)r * Nn] = f32_to_bf16(acc[mi][ni][r]);
      }
    }
  }
}

// ---------------- el/er: per-row dots with attention vector halves ----------------
__global__ __launch_bounds__(128) void rowdots_kernel(const float* __restrict__ Wh,
                                                      const float* __restrict__ avec,
                                                      float* __restrict__ el, float* __restrict__ er) {
  const int wave = threadIdx.x >> 5;
  const int lane = threadIdx.x & 31;
  const int row = blockIdx.x * 4 + wave;
  const float* w = Wh + (size_t)row * GD;
  float s0 = 0.0f, s1 = 0.0f;
  for (int k = lane; k < GD; k += 32) {
    float v = w[k];
    s0 += v * avec[k];
    s1 += v * avec[GD + k];
  }
  #pragma unroll
  for (int off = 16; off; off >>= 1) {
    s0 += __shfl_down(s0, off, 32);
    s1 += __shfl_down(s1, off, 32);
  }
  if (lane == 0) { el[row] = s0; er[row] = s1; }
}

// ---------------- masked leaky-relu attention + row softmax -> bf16 ----------------
__global__ __launch_bounds__(256) void att_softmax_kernel(const float* __restrict__ Aadj,
                                                          const float* __restrict__ el,
                                                          const float* __restrict__ er,
                                                          unsigned short* __restrict__ attbf) {
  __shared__ float red[8];
  const int i = blockIdx.x;
  const int tid = threadIdx.x;
  const int lane = tid & 31;
  const int wave = tid >> 5;
  const float eli = el[i];

  float vals[8];
  float vmax = -3.0e38f;
  #pragma unroll
  for (int t = 0; t < 8; ++t) {
    int j = tid + t * 256;
    float a = Aadj[(size_t)i * GN + j];
    float e = eli + er[j];
    e = (e > 0.0f) ? e : 0.2f * e;              // leaky_relu(0.2)
    float v = (a > 0.0f) ? a * e : -9.0e15f;
    vals[t] = v;
    vmax = fmaxf(vmax, v);
  }
  #pragma unroll
  for (int off = 16; off; off >>= 1) vmax = fmaxf(vmax, __shfl_xor(vmax, off, 32));
  if (lane == 0) red[wave] = vmax;
  __syncthreads();
  float bmax = red[0];
  #pragma unroll
  for (int w = 1; w < 8; ++w) bmax = fmaxf(bmax, red[w]);
  __syncthreads();

  float sum = 0.0f;
  float ev[8];
  #pragma unroll
  for (int t = 0; t < 8; ++t) { ev[t] = __expf(vals[t] - bmax); sum += ev[t]; }
  #pragma unroll
  for (int off = 16; off; off >>= 1) sum += __shfl_xor(sum, off, 32);
  if (lane == 0) red[wave] = sum;
  __syncthreads();
  float total = red[0];
  #pragma unroll
  for (int w = 1; w < 8; ++w) total += red[w];
  float inv = 1.0f / total;
  #pragma unroll
  for (int t = 0; t < 8; ++t)
    attbf[(size_t)i * GN + tid + t * 256] = f32_to_bf16(ev[t] * inv);
}

// ---------------- driver ----------------
extern "C" void kernel_launch(void* const* d_in, const int* in_sizes, int n_in,
                              void* d_out, int out_size, void* d_ws, size_t ws_size,
                              hipStream_t stream) {
  const float* X      = (const float*)d_in[1];
  const float* gumbel = (const float*)d_in[2];
  const float* a_link = (const float*)d_in[3];
  const float* Ws0    = (const float*)d_in[4];
  const float* Ws1    = (const float*)d_in[5];
  const float* a0     = (const float*)d_in[6];
  const float* a1     = (const float*)d_in[7];

  float* outH = (float*)d_out;                       // [N, D]
  float* outA = outH + (size_t)GN * GD;              // [N, N]
  float* outX = outA + (size_t)GN * GN;              // [N, F]

  char* ws = (char*)d_ws;
  size_t off = 0;
  auto take = [&](size_t bytes) -> char* {
    char* p = ws + off;
    off += (bytes + 255) & ~(size_t)255;
    return p;
  };
  unsigned short* Hbf   = (unsigned short*)take((size_t)GN * GF * 2);
  unsigned short* Wbf   = (unsigned short*)take((size_t)GF * GD * 2);
  float*          Whf   = (float*)take((size_t)GN * GD * 4);
  unsigned short* Whbf  = (unsigned short*)take((size_t)GN * GD * 2);
  unsigned short* attbf = (unsigned short*)take((size_t)GN * GN * 2);
  float*          Hnext = (float*)take((size_t)GN * GD * 4);
  float*          el    = (float*)take((size_t)GN * 4);
  float*          er    = (float*)take((size_t)GN * 4);

  // Adjacency (Aadj) straight into output slot, X passthrough
  linkpred_kernel<<<dim3(GN / 64, GN / 64), 256, 0, stream>>>(X, gumbel, a_link, outA);
  copy_f32_kernel<<<(GN * GF + 255) / 256, 256, 0, stream>>>(X, outX, GN * GF);

  // Layer 0 operand converts (layer 1's H arrives in bf16 via GEMM2's fused epilogue)
  cvt_bf16_kernel<<<(GN * GF + 255) / 256, 256, 0, stream>>>(X, Hbf, GN * GF);

  const float* Wk[2] = {Ws0, Ws1};
  const float* ak[2] = {a0, a1};
  float* Cout[2] = {Hnext, outH};

  for (int k = 0; k < 2; ++k) {
    cvt_bf16_kernel<<<(GF * GD + 255) / 256, 256, 0, stream>>>(Wk[k], Wbf, GF * GD);
    // Wh = H @ W   (M=2048, N=512, K=512); fused bf16 copy of Wh for GEMM2's B operand
    gemm_bf16_wmma_kernel<<<dim3(GD / NBLK, GN / MBLK), 256, 0, stream>>>(
        Hbf, Wbf, Whf, Whbf, GN, GD, GF);
    rowdots_kernel<<<GN / 4, 128, 0, stream>>>(Whf, ak[k], el, er);
    att_softmax_kernel<<<GN, 256, 0, stream>>>(outA, el, er, attbf);
    // H' = att @ Wh (M=2048, N=512, K=2048); layer 0 also emits H' in bf16 as
    // next layer's GEMM1 A-operand, layer 1 writes f32 H straight to d_out
    gemm_bf16_wmma_kernel<<<dim3(GD / NBLK, GN / MBLK), 256, 0, stream>>>(
        attbf, Whbf, Cout[k], (k == 0) ? Hbf : (unsigned short*)nullptr, GN, GD, GN);
  }
}
